// PeriodFDV3_15633680957969
// MI455X (gfx1250) — compile-verified, matrix operand order checked by reading, not run on hardware
//
#include <hip/hip_runtime.h>

typedef __bf16 bf16;
typedef __bf16 v16bf __attribute__((ext_vector_type(16)));
typedef __bf16 v8bf  __attribute__((ext_vector_type(8)));
typedef float  v8f   __attribute__((ext_vector_type(8)));
typedef int    v4i_  __attribute__((vector_size(4 * sizeof(int))));

#define NB    32
#define NL    720
#define ND    862
#define NS    60
#define NBD   (NB * ND)        /* 27584 */
#define MPAD  27648            /* 216 * 128 */
#define K720P 736              /* 23 * 32  */
#define N720P 768              /* 6 * 128  */
#define PI_F  3.14159265358979323846f

// gfx1250 async global->LDS copy (ASYNCcnt-tracked), with safe fallback.
// Builtin signature (from hipcc diagnostic): (int4 AS1*, int4 AS3*, imm, imm).
#if defined(__gfx1250__) && __has_builtin(__builtin_amdgcn_global_load_async_to_lds_b128) && __has_builtin(__builtin_amdgcn_s_wait_asynccnt)
#define USE_ASYNC_LDS 1
#define GASYNC_B128(gp, lp)                                                  \
  __builtin_amdgcn_global_load_async_to_lds_b128(                            \
      (__attribute__((address_space(1))) v4i_*)(gp),                         \
      (__attribute__((address_space(3))) v4i_*)(lp), 0, 0)
#else
#define USE_ASYNC_LDS 0
#endif

static __device__ __forceinline__ v8f v8f_zero() {
  v8f z;
#pragma unroll
  for (int i = 0; i < 8; ++i) z[i] = 0.f;
  return z;
}

// ---------------------------------------------------------------- zero scratch
__global__ void k_zero(uint4* __restrict__ p, long n16) {
  long i = (long)blockIdx.x * blockDim.x + threadIdx.x;
  const long stride = (long)gridDim.x * blockDim.x;
  uint4 z; z.x = z.y = z.z = z.w = 0u;
  for (; i < n16; i += stride) p[i] = z;
}

// ------------------------------------------------------- mean over time axis L
__global__ void k_mean_all(const float* __restrict__ x, float* __restrict__ ma) {
  int i = blockIdx.x * blockDim.x + threadIdx.x;   // over B*D
  if (i >= NBD) return;
  int b = i / ND, d = i % ND;
  const float* p = x + (long)b * NL * ND + d;
  float s = 0.f;
  for (int t = 0; t < NL; ++t) s += p[(long)t * ND];
  ma[i] = s * (1.f / NL);
}

// ------------------- per-period mean/std, normalized series, MLP "in" operands
__global__ void k_stats(const float* __restrict__ x, const float* __restrict__ ma,
                        float* __restrict__ normT, bf16* __restrict__ normT_bf,
                        bf16* __restrict__ AinM, bf16* __restrict__ AinS) {
  long i = (long)blockIdx.x * blockDim.x + threadIdx.x;  // over B*S*D, d fastest
  if (i >= (long)NB * NS * ND) return;
  int d = (int)(i % ND);
  int s = (int)((i / ND) % NS);
  int b = (int)(i / ((long)ND * NS));
  const float* p = x + ((long)b * NL + s * 12) * ND + d;
  float v[12]; float m = 0.f;
#pragma unroll
  for (int j = 0; j < 12; ++j) { v[j] = p[(long)j * ND]; m += v[j]; }
  m *= (1.f / 12.f);
  float var = 0.f;
#pragma unroll
  for (int j = 0; j < 12; ++j) { float t = v[j] - m; var += t * t; }
  float sd = sqrtf(var * (1.f / 11.f));            // ddof = 1
  long bd = (long)b * ND + d;
  AinM[bd * 64 + s] = (bf16)(m - ma[bd]);
  AinS[bd * 64 + s] = (bf16)sd;
  float inv = 1.f / (sd + 1e-8f);
#pragma unroll
  for (int j = 0; j < 12; ++j) {
    float nv = (v[j] - m) * inv;
    normT[bd * (long)NL + s * 12 + j] = nv;
    normT_bf[bd * (long)K720P + s * 12 + j] = (bf16)nv;
  }
}

// ------------------------------------------------- raw-series GEMM A operands
__global__ void k_raw(const float* __restrict__ x, const float* __restrict__ ma,
                      bf16* __restrict__ ArawM, bf16* __restrict__ ArawS) {
  long i = (long)blockIdx.x * blockDim.x + threadIdx.x;  // over B*L*D, d fastest
  if (i >= (long)NB * NL * ND) return;
  int d = (int)(i % ND);
  int t = (int)((i / ND) % NL);
  int b = (int)(i / ((long)ND * NL));
  float v = x[i];
  long bd = (long)b * ND + d;
  float mav = ma[bd];
  ArawM[bd * (long)K720P + t] = (bf16)(v - mav);
  ArawS[bd * (long)K720P + t] = (bf16)v;
}

// --------------------------- DFT basis, stored TRANSPOSED: BT[f][t] (Npad x Kpad)
__global__ void k_dft(bf16* __restrict__ CcosT, bf16* __restrict__ CsinT) {
  int i = blockIdx.x * blockDim.x + threadIdx.x;   // over N720P*K720P
  if (i >= N720P * K720P) return;
  int t = i % K720P, f = i / K720P;
  float c = 0.f, sn = 0.f;
  if (t < NL && f < NL) {
    int r = (int)(((long)t * f) % NL);             // exact angle reduction
    float ang = (2.0f * PI_F / NL) * (float)r;
    c = __cosf(ang);
    sn = -__sinf(ang);                             // Im part of e^{-i2pi t f/L}
  }
  CcosT[i] = (bf16)c;
  CsinT[i] = (bf16)sn;
}

// ---------------- weight -> bf16, stored TRANSPOSED: BT[n][k] = w[n][k] (padded)
__global__ void k_wtT(const float* __restrict__ w, bf16* __restrict__ BT,
                      int N, int K, int Kpad, int Npad) {
  int i = blockIdx.x * blockDim.x + threadIdx.x;   // over Npad*Kpad
  if (i >= Npad * Kpad) return;
  int k = i % Kpad, n = i / Kpad;
  float v = (k < K && n < N) ? w[(long)n * K + k] : 0.f;
  BT[i] = (bf16)v;
}

// ------------------------------------------------------------ WMMA bf16 GEMM
// C[Mpad x Npad](f32) = A[Mpad x Kpad](bf16 row-major) * BT[Npad x Kpad]^T
// Compile-time KSTEPS/LDK/LDC. Block tile 128x128, 8 waves (wave32), K-step 32.
// Double-buffered LDS filled by GLOBAL_LOAD_ASYNC_TO_LDS_B128 (ASYNCcnt) when
// available; register-staged fallback otherwise. No bounds checks (padded).
template <int KSTEPS, int LDK, int LDC>
__global__ __launch_bounds__(256)
void k_gemm_bf16(const bf16* __restrict__ A, const bf16* __restrict__ BT,
                 float* __restrict__ C) {
#if USE_ASYNC_LDS
  __shared__ bf16 As[2][128][40];   // +8 pad: kill LDS bank conflicts
  __shared__ bf16 Bs[2][128][40];
#else
  __shared__ bf16 As[1][128][40];
  __shared__ bf16 Bs[1][128][40];
#endif

  const int tid  = threadIdx.x;
  const int lane = tid & 31;
  const int wid  = tid >> 5;
  const int wm   = wid >> 2;                 // 0..1 : 64-row strip
  const int wn   = wid & 3;                  // 0..3 : 32-col strip
  const long rowBase = (long)blockIdx.y * 128;
  const long colBase = (long)blockIdx.x * 128;

  const int tr = tid >> 1;                   // tile row 0..127 (both tiles)
  const int tc = (tid & 1) << 4;             // K offset 0/16

  const bf16* aRow = A  + (rowBase + tr) * (long)LDK + tc;
  const bf16* bRow = BT + (colBase + tr) * (long)LDK + tc;

  v8f acc[4][2];
#pragma unroll
  for (int mi = 0; mi < 4; ++mi)
#pragma unroll
    for (int ni = 0; ni < 2; ++ni) acc[mi][ni] = v8f_zero();

  const int hlf = lane >> 4;                 // ISA 16-bit fragment layout
  const int l15 = lane & 15;
  const int kh  = hlf << 3;                  // K-halves 0..7 / 8..15 (+16 hi)

#if USE_ASYNC_LDS
  // prologue: async-DMA tile 0 into buffer 0
  GASYNC_B128(aRow,     &As[0][tr][tc]);
  GASYNC_B128(aRow + 8, &As[0][tr][tc + 8]);
  GASYNC_B128(bRow,     &Bs[0][tr][tc]);
  GASYNC_B128(bRow + 8, &Bs[0][tr][tc + 8]);
  __builtin_amdgcn_s_wait_asynccnt(0);
  __syncthreads();
#endif

  for (int ks = 0; ks < KSTEPS; ++ks) {
#if USE_ASYNC_LDS
    const int cur = ks & 1;
    if (ks + 1 < KSTEPS) {                   // prefetch next tile (ping-pong)
      const int k1 = (ks + 1) << 5;
      GASYNC_B128(aRow + k1,     &As[cur ^ 1][tr][tc]);
      GASYNC_B128(aRow + k1 + 8, &As[cur ^ 1][tr][tc + 8]);
      GASYNC_B128(bRow + k1,     &Bs[cur ^ 1][tr][tc]);
      GASYNC_B128(bRow + k1 + 8, &Bs[cur ^ 1][tr][tc + 8]);
    }
#else
    const int cur = 0;
    const int k0 = ks << 5;
    uint4 a0 = *(const uint4*)(aRow + k0);
    uint4 a1 = *(const uint4*)(aRow + k0 + 8);
    uint4 b0 = *(const uint4*)(bRow + k0);
    uint4 b1 = *(const uint4*)(bRow + k0 + 8);
    __syncthreads();
    *(uint4*)&As[0][tr][tc]     = a0;
    *(uint4*)&As[0][tr][tc + 8] = a1;
    *(uint4*)&Bs[0][tr][tc]     = b0;
    *(uint4*)&Bs[0][tr][tc + 8] = b1;
    __syncthreads();
#endif

    v16bf bfrag[2];
#pragma unroll
    for (int ni = 0; ni < 2; ++ni) {
      const int col = (wn << 5) + (ni << 4) + l15;
      v8bf lo = *(const v8bf*)&Bs[cur][col][kh];
      v8bf hi = *(const v8bf*)&Bs[cur][col][kh + 16];
      bfrag[ni] = __builtin_shufflevector(lo, hi,
          0, 1, 2, 3, 4, 5, 6, 7, 8, 9, 10, 11, 12, 13, 14, 15);
    }
#pragma unroll
    for (int mi = 0; mi < 4; ++mi) {
      const int row = (wm << 6) + (mi << 4) + l15;
      v8bf lo = *(const v8bf*)&As[cur][row][kh];
      v8bf hi = *(const v8bf*)&As[cur][row][kh + 16];
      v16bf av = __builtin_shufflevector(lo, hi,
          0, 1, 2, 3, 4, 5, 6, 7, 8, 9, 10, 11, 12, 13, 14, 15);
#pragma unroll
      for (int ni = 0; ni < 2; ++ni) {
        acc[mi][ni] = __builtin_amdgcn_wmma_f32_16x16x32_bf16(
            false, av, false, bfrag[ni], (short)0, acc[mi][ni], false, false);
      }
    }

#if USE_ASYNC_LDS
    __builtin_amdgcn_s_wait_asynccnt(0);     // next tile landed
    __syncthreads();                          // everyone done reading cur
#endif
  }

  // C/D layout: VGPR r -> M = r + 8*(lane>=16), N = lane&15
#pragma unroll
  for (int mi = 0; mi < 4; ++mi) {
    const long row0 = rowBase + (wm << 6) + (mi << 4) + (hlf << 3);
#pragma unroll
    for (int ni = 0; ni < 2; ++ni) {
      const long col = colBase + (wn << 5) + (ni << 4) + l15;
      float* Cp = C + row0 * (long)LDC + col;
#pragma unroll
      for (int r = 0; r < 8; ++r) Cp[r * LDC] = acc[mi][ni][r];
    }
  }
}

// ------------------------------------------- top-30 |X(f)| per (b,d), one wave
__global__ void k_topk(const float* __restrict__ Xre, const float* __restrict__ Xim,
                       int* __restrict__ topkIdx) {
  int bd = blockIdx.x * (blockDim.x >> 5) + (threadIdx.x >> 5);
  int lane = threadIdx.x & 31;
  if (bd >= NBD) return;
  const float* re = Xre + (long)bd * N720P;
  const float* im = Xim + (long)bd * N720P;
  float mag[23];
#pragma unroll
  for (int j = 0; j < 23; ++j) {
    int f = lane + j * 32;
    mag[j] = (f < NL) ? (re[f] * re[f] + im[f] * im[f]) : -1.f;
  }
  for (int r = 0; r < 30; ++r) {
    float best = -1.f; int bj = 0;
#pragma unroll
    for (int j = 0; j < 23; ++j)
      if (mag[j] > best) { best = mag[j]; bj = j; }
    int bidx = lane + bj * 32;
    for (int off = 16; off > 0; off >>= 1) {
      float ob = __shfl_xor(best, off, 32);
      int   oi = __shfl_xor(bidx, off, 32);
      if (ob > best || (ob == best && oi < bidx)) { best = ob; bidx = oi; }
    }
    if (lane == 0) topkIdx[(long)bd * 30 + r] = bidx;
#pragma unroll
    for (int j = 0; j < 23; ++j)
      if (lane + j * 32 == bidx) mag[j] = -1.f;
  }
}

// --------- reconstruct x_f from 30 freqs; residual -> out rows [0,720); bf16 x_f
__global__ void k_recon(const float* __restrict__ Xre, const float* __restrict__ Xim,
                        const int* __restrict__ topkIdx, const float* __restrict__ normT,
                        float* __restrict__ out, bf16* __restrict__ xfT) {
  int bd = blockIdx.x;
  int b = bd / ND, d = bd % ND;
  __shared__ int   fs[30];
  __shared__ float fre[30], fim[30];
  if (threadIdx.x < 30) {
    int f = topkIdx[(long)bd * 30 + threadIdx.x];
    fs[threadIdx.x]  = f;
    fre[threadIdx.x] = Xre[(long)bd * N720P + f];
    fim[threadIdx.x] = Xim[(long)bd * N720P + f];
  }
  __syncthreads();
  for (int t = threadIdx.x; t < NL; t += blockDim.x) {
    float s = 0.f;
#pragma unroll
    for (int r = 0; r < 30; ++r) {
      int rem = (int)(((long)fs[r] * t) % NL);
      float ang = (2.f * PI_F / NL) * (float)rem;
      s += fre[r] * __cosf(ang) - fim[r] * __sinf(ang);
    }
    s *= (1.f / NL);
    float nv = normT[(long)bd * NL + t];
    out[((long)b * 1560 + t) * ND + d] = nv - s;   // residual
    xfT[(long)bd * K720P + t] = (bf16)s;
  }
}

// ----------------------------------- hidden activation: bias + tanh/relu -> bf16
__global__ void k_act(const float* __restrict__ Htmp, const float* __restrict__ b_in,
                      const float* __restrict__ b_raw, bf16* __restrict__ Hbf, int mode) {
  long i = (long)blockIdx.x * blockDim.x + threadIdx.x;  // over MPAD*1024
  if (i >= (long)MPAD * 1024) return;
  int n = (int)(i & 1023);
  float bias = (n < 512) ? b_in[n] : b_raw[n - 512];
  float v = Htmp[i] + bias;
  v = mode ? fmaxf(v, 0.f) : tanhf(v);
  Hbf[i] = (bf16)v;
}

// ---------------------------------------- freq-MLP hidden: bias + relu -> bf16
__global__ void k_factm(const float* __restrict__ Hf32, const float* __restrict__ b1,
                        bf16* __restrict__ Hfbf) {
  long i = (long)blockIdx.x * blockDim.x + threadIdx.x;  // over MPAD*64
  if (i >= (long)MPAD * 64) return;
  int n = (int)(i & 63);
  long m = i >> 6;
  float v = Hf32[m * 128 + n] + b1[n];
  Hfbf[i] = (bf16)fmaxf(v, 0.f);
}

// ----------------------------------------- pred_main -> out rows [720, 1440)
__global__ void k_pred_main(const float* __restrict__ Pf, const float* __restrict__ b2,
                            float* __restrict__ out) {
  long i = (long)blockIdx.x * blockDim.x + threadIdx.x;  // over NBD*720
  if (i >= (long)NBD * NL) return;
  int t = (int)(i % NL);
  long m = i / NL;
  int b = (int)(m / ND), d = (int)(m % ND);
  out[((long)b * 1560 + 720 + t) * ND + d] = Pf[m * N720P + t] + b2[t];
}

// ---------------------------------------- preds_mean -> out rows [1440, 1500)
__global__ void k_pred_mean(const float* __restrict__ Om, const float* __restrict__ bo,
                            const float* __restrict__ weight, const float* __restrict__ ma,
                            float* __restrict__ out) {
  long i = (long)blockIdx.x * blockDim.x + threadIdx.x;  // over NBD*60
  if (i >= (long)NBD * NS) return;
  int s = (int)(i % NS);
  long m = i / NS;
  int b = (int)(m / ND), d = (int)(m % ND);
  float v = Om[m * 128 + s] + bo[s];
  v = v * weight[d] + ma[m] * weight[ND + d];
  out[((long)b * 1560 + 1440 + s) * ND + d] = v;
}

// ----------------------------------------- preds_std -> out rows [1500, 1560)
__global__ void k_pred_std(const float* __restrict__ Os, const float* __restrict__ bo,
                           float* __restrict__ out) {
  long i = (long)blockIdx.x * blockDim.x + threadIdx.x;  // over NBD*60
  if (i >= (long)NBD * NS) return;
  int s = (int)(i % NS);
  long m = i / NS;
  int b = (int)(m / ND), d = (int)(m % ND);
  float v = fmaxf(Os[m * 128 + s] + bo[s], 0.f);
  out[((long)b * 1560 + 1500 + s) * ND + d] = v;
}

// ============================================================================
extern "C" void kernel_launch(void* const* d_in, const int* in_sizes, int n_in,
                              void* d_out, int out_size, void* d_ws, size_t ws_size,
                              hipStream_t stream) {
  (void)in_sizes; (void)n_in; (void)out_size; (void)ws_size;
  const float* batch_x = (const float*)d_in[0];
  const float* w_in_m  = (const float*)d_in[1];
  const float* b_in_m  = (const float*)d_in[2];
  const float* w_raw_m = (const float*)d_in[3];
  const float* b_raw_m = (const float*)d_in[4];
  const float* w_out_m = (const float*)d_in[5];
  const float* b_out_m = (const float*)d_in[6];
  const float* w_in_s  = (const float*)d_in[7];
  const float* b_in_s  = (const float*)d_in[8];
  const float* w_raw_s = (const float*)d_in[9];
  const float* b_raw_s = (const float*)d_in[10];
  const float* w_out_s = (const float*)d_in[11];
  const float* b_out_s = (const float*)d_in[12];
  const float* w_f1    = (const float*)d_in[13];
  const float* b_f1    = (const float*)d_in[14];
  const float* w_f2    = (const float*)d_in[15];
  const float* b_f2    = (const float*)d_in[16];
  const float* weight  = (const float*)d_in[17];
  float* out = (float*)d_out;
  char*  ws  = (char*)d_ws;

  size_t off = 0;
  auto alloc = [&](size_t bytes) {
    size_t o = off; off += (bytes + 255) & ~(size_t)255; return o;
  };
  const size_t szAbig = (size_t)MPAD * K720P * 2;
  const size_t szAin  = (size_t)MPAD * 64 * 2;

  size_t o_normT_bf = alloc(szAbig);
  size_t o_ArawM    = alloc(szAbig);
  size_t o_ArawS    = alloc(szAbig);
  size_t o_AinM     = alloc(szAin);
  size_t o_AinS     = alloc(szAin);
  size_t o_xfT      = alloc(szAbig);
  size_t o_normT    = alloc((size_t)NBD * NL * 4);
  size_t o_Ccos     = alloc((size_t)N720P * K720P * 2);   // BT: [f][t]
  size_t o_Csin     = alloc((size_t)N720P * K720P * 2);
  size_t o_BinM     = alloc((size_t)512 * 64 * 2);        // BT: [n][k]
  size_t o_BinS     = alloc((size_t)512 * 64 * 2);
  size_t o_BrawM    = alloc((size_t)512 * K720P * 2);
  size_t o_BrawS    = alloc((size_t)512 * K720P * 2);
  size_t o_BoutM    = alloc((size_t)128 * 1024 * 2);
  size_t o_BoutS    = alloc((size_t)128 * 1024 * 2);
  size_t o_Bf1      = alloc((size_t)128 * K720P * 2);
  size_t o_Bf2      = alloc((size_t)N720P * 64 * 2);
  size_t o_Hbf      = alloc((size_t)MPAD * 1024 * 2);
  size_t o_Hfbf     = alloc((size_t)MPAD * 64 * 2);
  size_t o_meanAll  = alloc((size_t)NBD * 4);
  size_t o_topk     = alloc((size_t)NBD * 30 * 4);
  size_t o_big      = alloc((size_t)MPAD * N720P * 4 * 2);  // Xre|Xim region
  size_t o_Xre  = o_big;
  size_t o_Xim  = o_big + (size_t)MPAD * N720P * 4;
  size_t o_HfC  = o_Xim;                              // alias (after recon)
  size_t o_Pf   = o_Xre;                              // alias (after recon)
  size_t o_Htmp = o_Xre;                              // alias (after pred_main)
  size_t o_O    = o_Xre + (size_t)MPAD * 1024 * 4;    // alias tail of region

  dim3 blk(256);

  auto zero = [&](size_t o, size_t bytes) {
    k_zero<<<dim3(4096), blk, 0, stream>>>((uint4*)(ws + o), (long)(bytes / 16));
  };
  zero(o_normT_bf, szAbig);
  zero(o_ArawM, szAbig);
  zero(o_ArawS, szAbig);
  zero(o_AinM, szAin);
  zero(o_AinS, szAin);
  zero(o_xfT, szAbig);

  k_mean_all<<<dim3((NBD + 255) / 256), blk, 0, stream>>>(batch_x, (float*)(ws + o_meanAll));

  long nStats = (long)NB * NS * ND;
  k_stats<<<dim3((unsigned)((nStats + 255) / 256)), blk, 0, stream>>>(
      batch_x, (float*)(ws + o_meanAll), (float*)(ws + o_normT),
      (bf16*)(ws + o_normT_bf), (bf16*)(ws + o_AinM), (bf16*)(ws + o_AinS));

  long nRaw = (long)NB * NL * ND;
  k_raw<<<dim3((unsigned)((nRaw + 255) / 256)), blk, 0, stream>>>(
      batch_x, (float*)(ws + o_meanAll), (bf16*)(ws + o_ArawM), (bf16*)(ws + o_ArawS));

  k_dft<<<dim3((N720P * K720P + 255) / 256), blk, 0, stream>>>(
      (bf16*)(ws + o_Ccos), (bf16*)(ws + o_Csin));

  auto wt = [&](const float* w, size_t o, int N, int K, int Kpad, int Npad) {
    k_wtT<<<dim3((Npad * Kpad + 255) / 256), blk, 0, stream>>>(
        w, (bf16*)(ws + o), N, K, Kpad, Npad);
  };
  wt(w_in_m,  o_BinM,  512, 60,   64,    512);
  wt(w_in_s,  o_BinS,  512, 60,   64,    512);
  wt(w_raw_m, o_BrawM, 512, 720,  K720P, 512);
  wt(w_raw_s, o_BrawS, 512, 720,  K720P, 512);
  wt(w_out_m, o_BoutM, 60,  1024, 1024,  128);
  wt(w_out_s, o_BoutS, 60,  1024, 1024,  128);
  wt(w_f1,    o_Bf1,   64,  720,  K720P, 128);
  wt(w_f2,    o_Bf2,   720, 64,   64,    N720P);

#define GEMM(KS, LDKv, LDCv, oA, oBT, oC, Npad)                               \
  k_gemm_bf16<KS, LDKv, LDCv><<<dim3((Npad) / 128, MPAD / 128), blk, 0,       \
      stream>>>((const bf16*)(ws + (oA)), (const bf16*)(ws + (oBT)),          \
                (float*)(ws + (oC)))

  // ---- FFT branch: DFT GEMMs, top-k, reconstruction, freq MLP
  GEMM(23, K720P, N720P, o_normT_bf, o_Ccos, o_Xre, N720P);
  GEMM(23, K720P, N720P, o_normT_bf, o_Csin, o_Xim, N720P);

  k_topk<<<dim3((NBD + 7) / 8), blk, 0, stream>>>(
      (const float*)(ws + o_Xre), (const float*)(ws + o_Xim), (int*)(ws + o_topk));
  k_recon<<<dim3(NBD), blk, 0, stream>>>(
      (const float*)(ws + o_Xre), (const float*)(ws + o_Xim), (const int*)(ws + o_topk),
      (const float*)(ws + o_normT), out, (bf16*)(ws + o_xfT));

  GEMM(23, K720P, 128, o_xfT, o_Bf1, o_HfC, 128);
  k_factm<<<dim3((unsigned)(((long)MPAD * 64 + 255) / 256)), blk, 0, stream>>>(
      (const float*)(ws + o_HfC), b_f1, (bf16*)(ws + o_Hfbf));
  GEMM(2, 64, N720P, o_Hfbf, o_Bf2, o_Pf, N720P);
  k_pred_main<<<dim3((unsigned)(((long)NBD * NL + 255) / 256)), blk, 0, stream>>>(
      (const float*)(ws + o_Pf), b_f2, out);

  // ---- mean branch
  GEMM(2,  64,    1024, o_AinM,  o_BinM,  o_Htmp,           512);
  GEMM(23, K720P, 1024, o_ArawM, o_BrawM, o_Htmp + 512 * 4, 512);
  k_act<<<dim3((unsigned)(((long)MPAD * 1024 + 255) / 256)), blk, 0, stream>>>(
      (const float*)(ws + o_Htmp), b_in_m, b_raw_m, (bf16*)(ws + o_Hbf), 0);
  GEMM(32, 1024, 128, o_Hbf, o_BoutM, o_O, 128);
  k_pred_mean<<<dim3((unsigned)(((long)NBD * NS + 255) / 256)), blk, 0, stream>>>(
      (const float*)(ws + o_O), b_out_m, weight, (const float*)(ws + o_meanAll), out);

  // ---- std branch
  GEMM(2,  64,    1024, o_AinS,  o_BinS,  o_Htmp,           512);
  GEMM(23, K720P, 1024, o_ArawS, o_BrawS, o_Htmp + 512 * 4, 512);
  k_act<<<dim3((unsigned)(((long)MPAD * 1024 + 255) / 256)), blk, 0, stream>>>(
      (const float*)(ws + o_Htmp), b_in_s, b_raw_s, (bf16*)(ws + o_Hbf), 1);
  GEMM(32, 1024, 128, o_Hbf, o_BoutS, o_O, 128);
  k_pred_std<<<dim3((unsigned)(((long)NBD * NS + 255) / 256)), blk, 0, stream>>>(
      (const float*)(ws + o_O), b_out_s, out);
#undef GEMM
}
